// MSA_24051816858266
// MI455X (gfx1250) — compile-verified, matrix operand order checked by reading, not run on hardware
//
#include <hip/hip_runtime.h>
#include <hip/hip_bf16.h>

// ---------------------------------------------------------------------------
// Multi-head self-attention forward for MI455X (gfx1250, wave32, WMMA).
//   B=2, N=2048, H=12, Dh=64, DIM_IN=768, DQKV=2304, SCALE=1/8
// Pipeline:
//   1) qkv_gemm : X[4096,768]f32 @ Wqkv[768,2304]f32 -> Q,K (f16 [B,H,N,D]),
//                 V (f16, pre-transposed [B,H,D,N]). Q pre-scaled by 1/8.
//                 Double-buffered LDS, one barrier per k-step.
//   2) attn_fwd : flash-attention, 64 query rows / block, online softmax.
//                 K/V tiles prefetched with GLOBAL_LOAD_ASYNC_TO_LDS_B128
//                 (ASYNCcnt) into double-buffered LDS.
//   3) out_gemm : heads[4096,768]f16 @ Wout[768,768] + bias -> f32 out.
// ---------------------------------------------------------------------------

typedef _Float16 v8h  __attribute__((ext_vector_type(8)));
typedef _Float16 v16h __attribute__((ext_vector_type(16)));
typedef float    v8f  __attribute__((ext_vector_type(8)));

#define NTOK   4096          // B*N
#define DIN    768
#define DQKV   2304
#define NHEAD  12
#define DH     64
#define SEQ    2048
#define HEADSZ ((size_t)SEQ * DH)                 // per (b,h) plane
#define QKVELEMS ((size_t)2 * NHEAD * SEQ * DH)   // 3,145,728

__device__ __forceinline__ v8f wmma_f16(v16h a, v16h b, v8f c) {
  return __builtin_amdgcn_wmma_f32_16x16x32_f16(
      /*neg_a=*/false, a, /*neg_b=*/false, b,
      /*c_mod=*/(short)0, c, /*reuse_a=*/false, /*reuse_b=*/false);
}

// Load a 16x32 f16 A-fragment (or 32x16 B-fragment, identical per-lane layout
// when the source is stored "contraction-contiguous"):
//   lanes 0-15 : row rc_base+lane, K = {0..7, 16..23} of the 32-chunk
//   lanes 16-31: row rc_base+lane-16, K = {8..15, 24..31}
__device__ __forceinline__ v16h load_frag(const _Float16* base, int stride,
                                          int rc_base, int k_base, int lane) {
  int rc = rc_base + (lane & 15);
  int kh = (lane >> 4) << 3;                 // 0 or 8
  const _Float16* p = base + rc * stride + k_base + kh;
  v8h lo = *(const v8h*)p;                   // K 0..7  (+kh)
  v8h hi = *(const v8h*)(p + 16);            // K 16..23 (+kh)
  return __builtin_shufflevector(lo, hi, 0, 1, 2, 3, 4, 5, 6, 7,
                                 8, 9, 10, 11, 12, 13, 14, 15);
}

// Async memory->LDS copy, 16B per lane. ASYNCcnt-tracked (cdna5 §08).
// Generic (flat) LDS address truncated to 32 bits == LDS offset per the
// aperture mapping; VDST carries the LDS byte address.
__device__ __forceinline__ void async_copy_b128(void* lds, const void* gptr) {
  unsigned l = (unsigned)(uintptr_t)lds;
  unsigned long long g = (unsigned long long)(uintptr_t)gptr;
  asm volatile("global_load_async_to_lds_b128 %0, %1, off"
               :: "v"(l), "v"(g)
               : "memory");
}

__device__ __forceinline__ void wait_async_le8() {
  asm volatile("s_wait_asynccnt 8" ::: "memory");
}
__device__ __forceinline__ void wait_async_0() {
  asm volatile("s_wait_asynccnt 0" ::: "memory");
}

// ---------------------------------------------------------------------------
// Kernel 1: QKV projection.  grid(32, 9), block 256 (8 waves, 2x4 wave grid)
// WG tile 128(M) x 256(N), BK=32, double-buffered LDS (one barrier/step).
// ---------------------------------------------------------------------------
__global__ __launch_bounds__(256) void qkv_gemm(
    const float* __restrict__ X, const float* __restrict__ W,
    _Float16* __restrict__ Qh, _Float16* __restrict__ Kh,
    _Float16* __restrict__ Vt) {
  __shared__ __align__(16) _Float16 As[2][128 * 40];  // [row][k], 80B stride
  __shared__ __align__(16) _Float16 Bs[2][256 * 40];  // [col][k] (transposed)

  const int t = threadIdx.x, lane = t & 31, w = t >> 5;
  const int wm = w >> 2, wn = w & 3;
  const int rowBase = blockIdx.x * 128;
  const int colBase = blockIdx.y * 256;

  auto stage = [&](int kk, int buf) {
    {  // A: 128x32 f32 -> f16
      const int c4 = t & 7, r0 = t >> 3;
#pragma unroll
      for (int p = 0; p < 4; ++p) {
        const int r = r0 + p * 32;
        const float4 f =
            *(const float4*)(X + (size_t)(rowBase + r) * DIN + kk + c4 * 4);
        _Float16* d = As[buf] + r * 40 + c4 * 4;
        d[0] = (_Float16)f.x; d[1] = (_Float16)f.y;
        d[2] = (_Float16)f.z; d[3] = (_Float16)f.w;
      }
    }
    {  // B transposed: Bs[n][k] = W[kk+k][colBase+n]
      const int nq = t & 63, k0 = t >> 6;
#pragma unroll
      for (int p = 0; p < 8; ++p) {
        const int k = k0 + p * 4;
        const float4 f =
            *(const float4*)(W + (size_t)(kk + k) * DQKV + colBase + nq * 4);
        _Float16* d = Bs[buf] + (nq * 4) * 40 + k;
        d[0]      = (_Float16)f.x;
        d[40]     = (_Float16)f.y;
        d[80]     = (_Float16)f.z;
        d[120]    = (_Float16)f.w;
      }
    }
  };

  v8f acc[4][4];
#pragma unroll
  for (int i = 0; i < 4; ++i)
#pragma unroll
    for (int j = 0; j < 4; ++j) acc[i][j] = {};

  stage(0, 0);
  for (int kk = 0, ib = 0; kk < DIN; kk += 32, ++ib) {
    const int buf = ib & 1;
    __syncthreads();
    if (kk + 32 < DIN) stage(kk + 32, buf ^ 1);

    v16h a[4];
#pragma unroll
    for (int i = 0; i < 4; ++i)
      a[i] = load_frag(As[buf], 40, wm * 64 + i * 16, 0, lane);
#pragma unroll
    for (int j = 0; j < 4; ++j) {
      const v16h b = load_frag(Bs[buf], 40, wn * 64 + j * 16, 0, lane);
#pragma unroll
      for (int i = 0; i < 4; ++i) acc[i][j] = wmma_f16(a[i], b, acc[i][j]);
    }
  }

  // Epilogue: scatter into Q (scaled), K, V(transposed).
#pragma unroll
  for (int i = 0; i < 4; ++i) {
#pragma unroll
    for (int j = 0; j < 4; ++j) {
#pragma unroll
      for (int v = 0; v < 8; ++v) {
        const int r = rowBase + wm * 64 + i * 16 + ((lane >> 4) << 3) + v;
        const int c = colBase + wn * 64 + j * 16 + (lane & 15);
        const int b = r >> 11, n = r & 2047;
        const int h = c / 192, rem = c % 192, s = rem >> 6, d = rem & 63;
        const float val = acc[i][j][v];
        const size_t bh = (size_t)(b * NHEAD + h);
        if (s == 0)
          Qh[(bh * SEQ + n) * DH + d] = (_Float16)(val * 0.125f);
        else if (s == 1)
          Kh[(bh * SEQ + n) * DH + d] = (_Float16)val;
        else
          Vt[(bh * DH + d) * SEQ + n] = (_Float16)val;
      }
    }
  }
}

// ---------------------------------------------------------------------------
// Kernel 2: flash attention. grid(N/64, B*H), block 128 (4 waves).
// Wave w owns query rows [w*16, w*16+16). 64-key blocks stream through
// double-buffered LDS via async global->LDS copies (ASYNCcnt).
// ---------------------------------------------------------------------------
__global__ __launch_bounds__(128) void attn_fwd(
    const _Float16* __restrict__ Qh, const _Float16* __restrict__ Kh,
    const _Float16* __restrict__ Vtg, _Float16* __restrict__ Hh) {
  __shared__ __align__(16) _Float16 Qs[64 * 80];          // [qrow][d]
  __shared__ __align__(16) _Float16 Ks[2][64 * 80];       // [key][d]
  __shared__ __align__(16) _Float16 Vs[2][64 * 80];       // [d][key]
  __shared__ __align__(16) _Float16 Ps[4 * 16 * 80];      // per-wave P tile

  const int t = threadIdx.x, lane = t & 31, w = t >> 5;
  const int bh = blockIdx.y;
  const int b = bh / NHEAD, h = bh % NHEAD;
  const int qbase = blockIdx.x * 64;

  const _Float16* Qg = Qh + ((size_t)bh * SEQ + qbase) * DH;
  const _Float16* Kg = Kh + (size_t)bh * HEADSZ;
  const _Float16* Vg = Vtg + (size_t)bh * HEADSZ;   // [d][n]

  const int c8 = t & 7, r0 = t >> 3;

  // stage Q once (plain f16 copy, 16B chunks)
  {
#pragma unroll
    for (int p = 0; p < 4; ++p) {
      const int r = r0 + p * 16;
      *(v8h*)(Qs + r * 80 + c8 * 8) = *(const v8h*)(Qg + r * DH + c8 * 8);
    }
  }

  // Issue one 64-key block's K and V tiles as 8 async b128s per thread.
  auto issue_block = [&](int jb, int buf) {
#pragma unroll
    for (int p = 0; p < 4; ++p) {
      const int r = r0 + p * 16;
      async_copy_b128(Ks[buf] + r * 80 + c8 * 8,
                      Kg + (size_t)(jb + r) * DH + c8 * 8);
      async_copy_b128(Vs[buf] + r * 80 + c8 * 8,
                      Vg + (size_t)r * SEQ + jb + c8 * 8);
    }
  };

  float mst[8], lst[8];
  v8f o[4];
#pragma unroll
  for (int v = 0; v < 8; ++v) { mst[v] = -1.0e30f; lst[v] = 0.0f; }
#pragma unroll
  for (int dt = 0; dt < 4; ++dt) o[dt] = {};

  _Float16* Pw = Ps + w * 16 * 80;

  issue_block(0, 0);
  for (int jb = 0, ib = 0; jb < SEQ; jb += 64, ++ib) {
    const int buf = ib & 1;
    __syncthreads();                 // buf^1 readers from iter ib-1 are done
    if (jb + 64 < SEQ) {
      issue_block(jb + 64, buf ^ 1); // prefetch next block (8 more asyncs)
      wait_async_le8();              // in-order: current block's 8 landed
    } else {
      wait_async_0();
    }
    __syncthreads();                 // every wave's current tile visible

    const _Float16* Kb = Ks[buf];
    const _Float16* Vb = Vs[buf];

    // S = Q K^T for this wave's 16 rows x 64 keys (contraction over Dh=64)
    const v16h aq0 = load_frag(Qs, 80, w * 16, 0, lane);
    const v16h aq1 = load_frag(Qs, 80, w * 16, 32, lane);
    v8f s[4];
#pragma unroll
    for (int jt = 0; jt < 4; ++jt) {
      const v16h b0 = load_frag(Kb, 80, jt * 16, 0, lane);
      const v16h b1 = load_frag(Kb, 80, jt * 16, 32, lane);
      v8f z = {};
      z = wmma_f16(aq0, b0, z);
      z = wmma_f16(aq1, b1, z);
      s[jt] = z;
    }

    // online softmax (rows live across 16-lane halves of the wave)
#pragma unroll
    for (int v = 0; v < 8; ++v) {
      const float x0 = s[0][v], x1 = s[1][v], x2 = s[2][v], x3 = s[3][v];
      float rm = fmaxf(fmaxf(x0, x1), fmaxf(x2, x3));
      rm = fmaxf(rm, __shfl_xor(rm, 1, 32));
      rm = fmaxf(rm, __shfl_xor(rm, 2, 32));
      rm = fmaxf(rm, __shfl_xor(rm, 4, 32));
      rm = fmaxf(rm, __shfl_xor(rm, 8, 32));
      const float mn = fmaxf(mst[v], rm);
      const float corr = __expf(mst[v] - mn);
      mst[v] = mn;
      const float p0 = __expf(x0 - mn), p1 = __expf(x1 - mn);
      const float p2 = __expf(x2 - mn), p3 = __expf(x3 - mn);
      float rs = p0 + p1 + p2 + p3;
      rs += __shfl_xor(rs, 1, 32);
      rs += __shfl_xor(rs, 2, 32);
      rs += __shfl_xor(rs, 4, 32);
      rs += __shfl_xor(rs, 8, 32);
      lst[v] = lst[v] * corr + rs;
      o[0][v] *= corr; o[1][v] *= corr; o[2][v] *= corr; o[3][v] *= corr;
      const int r = ((lane >> 4) << 3) + v, c = lane & 15;
      Pw[r * 80 + c]      = (_Float16)p0;
      Pw[r * 80 + 16 + c] = (_Float16)p1;
      Pw[r * 80 + 32 + c] = (_Float16)p2;
      Pw[r * 80 + 48 + c] = (_Float16)p3;
    }

    // O += P V (contraction over 64 keys; Vs is [d][key] so B-frags are
    // contiguous ds_load_b128)
    const v16h ap0 = load_frag(Pw, 80, 0, 0, lane);
    const v16h ap1 = load_frag(Pw, 80, 0, 32, lane);
#pragma unroll
    for (int dt = 0; dt < 4; ++dt) {
      const v16h b0 = load_frag(Vb, 80, dt * 16, 0, lane);
      const v16h b1 = load_frag(Vb, 80, dt * 16, 32, lane);
      o[dt] = wmma_f16(ap0, b0, o[dt]);
      o[dt] = wmma_f16(ap1, b1, o[dt]);
    }
  }

  // normalize and write heads buffer [B*N, H*Dh] in f16
#pragma unroll
  for (int v = 0; v < 8; ++v) {
    const float inv = 1.0f / lst[v];
    const int r = ((lane >> 4) << 3) + v;
    const int nrow = qbase + w * 16 + r;
    const size_t off =
        ((size_t)(b * SEQ + nrow)) * DIN + h * DH + (lane & 15);
    Hh[off + 0]  = (_Float16)(o[0][v] * inv);
    Hh[off + 16] = (_Float16)(o[1][v] * inv);
    Hh[off + 32] = (_Float16)(o[2][v] * inv);
    Hh[off + 48] = (_Float16)(o[3][v] * inv);
  }
}

// ---------------------------------------------------------------------------
// Kernel 3: output projection + bias. grid(32, 3), block 256,
// double-buffered LDS (one barrier per k-step).
// ---------------------------------------------------------------------------
__global__ __launch_bounds__(256) void out_gemm(
    const _Float16* __restrict__ Hh, const float* __restrict__ W,
    const float* __restrict__ bias, float* __restrict__ out) {
  __shared__ __align__(16) _Float16 As[2][128 * 40];
  __shared__ __align__(16) _Float16 Bs[2][256 * 40];

  const int t = threadIdx.x, lane = t & 31, w = t >> 5;
  const int wm = w >> 2, wn = w & 3;
  const int rowBase = blockIdx.x * 128;
  const int colBase = blockIdx.y * 256;

  auto stage = [&](int kk, int buf) {
    {  // A (already f16): 128x32, 16B chunks
      const int c8 = t & 3, r0 = t >> 2;
#pragma unroll
      for (int p = 0; p < 2; ++p) {
        const int r = r0 + p * 64;
        *(v8h*)(As[buf] + r * 40 + c8 * 8) =
            *(const v8h*)(Hh + (size_t)(rowBase + r) * DIN + kk + c8 * 8);
      }
    }
    {  // B transposed: Bs[n][k] = W[kk+k][colBase+n]
      const int nq = t & 63, k0 = t >> 6;
#pragma unroll
      for (int p = 0; p < 8; ++p) {
        const int k = k0 + p * 4;
        const float4 f =
            *(const float4*)(W + (size_t)(kk + k) * DIN + colBase + nq * 4);
        _Float16* d = Bs[buf] + (nq * 4) * 40 + k;
        d[0]   = (_Float16)f.x;
        d[40]  = (_Float16)f.y;
        d[80]  = (_Float16)f.z;
        d[120] = (_Float16)f.w;
      }
    }
  };

  v8f acc[4][4];
#pragma unroll
  for (int i = 0; i < 4; ++i)
#pragma unroll
    for (int j = 0; j < 4; ++j) acc[i][j] = {};

  stage(0, 0);
  for (int kk = 0, ib = 0; kk < DIN; kk += 32, ++ib) {
    const int buf = ib & 1;
    __syncthreads();
    if (kk + 32 < DIN) stage(kk + 32, buf ^ 1);

    v16h a[4];
#pragma unroll
    for (int i = 0; i < 4; ++i)
      a[i] = load_frag(As[buf], 40, wm * 64 + i * 16, 0, lane);
#pragma unroll
    for (int j = 0; j < 4; ++j) {
      const v16h b = load_frag(Bs[buf], 40, wn * 64 + j * 16, 0, lane);
#pragma unroll
      for (int i = 0; i < 4; ++i) acc[i][j] = wmma_f16(a[i], b, acc[i][j]);
    }
  }

#pragma unroll
  for (int i = 0; i < 4; ++i) {
#pragma unroll
    for (int j = 0; j < 4; ++j) {
#pragma unroll
      for (int v = 0; v < 8; ++v) {
        const int r = rowBase + wm * 64 + i * 16 + ((lane >> 4) << 3) + v;
        const int c = colBase + wn * 64 + j * 16 + (lane & 15);
        out[(size_t)r * DIN + c] = acc[i][j][v] + bias[c];
      }
    }
  }
}

// ---------------------------------------------------------------------------
extern "C" void kernel_launch(void* const* d_in, const int* in_sizes, int n_in,
                              void* d_out, int out_size, void* d_ws,
                              size_t ws_size, hipStream_t stream) {
  const float* x     = (const float*)d_in[0];  // [2,2048,768]
  const float* w_qkv = (const float*)d_in[1];  // [768,2304]
  const float* w_out = (const float*)d_in[2];  // [768,768]
  const float* b_out = (const float*)d_in[3];  // [768]
  float* out = (float*)d_out;                  // [2,2048,768]

  // workspace: Q, K (f16 [B,H,N,D]), Vt (f16 [B,H,D,N]), heads (f16 [4096,768])
  if (ws_size < 4 * QKVELEMS * sizeof(_Float16)) return;
  _Float16* Qh = (_Float16*)d_ws;
  _Float16* Kh = Qh + QKVELEMS;
  _Float16* Vt = Kh + QKVELEMS;
  _Float16* Hh = Vt + QKVELEMS;

  qkv_gemm<<<dim3(NTOK / 128, DQKV / 256), 256, 0, stream>>>(x, w_qkv, Qh, Kh,
                                                             Vt);
  attn_fwd<<<dim3(SEQ / 64, 2 * NHEAD), 128, 0, stream>>>(Qh, Kh, Vt, Hh);
  out_gemm<<<dim3(NTOK / 128, DIN / 256), 256, 0, stream>>>(Hh, w_out, b_out,
                                                            out);
}
